// MinGRUStack_75874892251384
// MI455X (gfx1250) — matrix operand order repaired and checked by reading, not compiled, not run on hardware
//
#include <hip/hip_runtime.h>
#include <math.h>

// ---- problem constants (match reference) ----
#define BB   8
#define TT   2048
#define INF  64
#define DD   512
#define OUTF 256
#define LL   4
#define PP   8
#define MM   (BB * TT)          // 16384 rows (tokens)

typedef __attribute__((ext_vector_type(16))) _Float16 v16h;
typedef __attribute__((ext_vector_type(8)))  _Float16 v8h;
typedef __attribute__((ext_vector_type(8)))  float    v8f;
typedef __attribute__((ext_vector_type(4)))  float    v4f;

// hat-function expansion of one x value onto the P=8 grid, as 8 f16s
__device__ __forceinline__ v8h basis8(float x) {
    const float t = (fminf(fmaxf(x, -1.f), 1.f) + 1.f) * 3.5f;  // 0.5*(P-1)
    v8h r;
    #pragma unroll
    for (int p = 0; p < 8; ++p)
        r[p] = (_Float16)fmaxf(1.f - fabsf(t - (float)p), 0.f);
    return r;
}

// -------------------------------------------------------------------------
// One-time weight conversion f32 -> f16 (each weight is reused M/64 = 256x
// per GEMM; convert once per launch instead of per use in the hot loop).
// -------------------------------------------------------------------------
__global__ __launch_bounds__(256)
void w_to_f16_kernel(const float* __restrict__ src, _Float16* __restrict__ dst,
                     int n8)   // n/8 vector chunks
{
    const int i = blockIdx.x * blockDim.x + threadIdx.x;
    if (i >= n8) return;
    const v4f a = *(const v4f*)(src + (size_t)i * 8);
    const v4f b = *(const v4f*)(src + (size_t)i * 8 + 4);
    v8h o;
    #pragma unroll
    for (int j = 0; j < 4; ++j) { o[j] = (_Float16)a[j]; o[4 + j] = (_Float16)b[j]; }
    *(v8h*)(dst + (size_t)i * 8) = o;
}

// -------------------------------------------------------------------------
// APL GEMM:  Y[m, o] = sum_{i,p} basis(X[m,i])[p] * W[i, o, p]
// GEMM with K = In*P.  Block = 256 threads = 8 waves, tile M=64 x N=128.
// Stage = 64 K-values (8 input features, two 32-K panels), double-buffered
// in LDS (16 KB), stored directly in WMMA A-fragment layout. Per stage:
// 256 threads expand 2 basis values each; each wave then runs 2 panels x
// (2 global_load_b128 f16 B + 4 ds_load'ed A frags -> 4 WMMA) = 8 WMMAs
// per barrier. Software-pipelined: no branches in the steady-state loop.
// -------------------------------------------------------------------------
__global__ __launch_bounds__(256)
void apl_gemm_kernel(const float* __restrict__ X,       // (M, In) f32
                     const _Float16* __restrict__ W,    // (In, Out, P) f16
                     float* __restrict__ Y,             // (M, Out) f32
                     int In, int Out)
{
    // [buf][panel][(subtile*32 + lane)*16 + elem] ; 2 x 2 x 4KB
    __shared__ __attribute__((aligned(32))) _Float16 ldsA[2][2][4 * 32 * 16];

    const int tid    = threadIdx.x;
    const int lane   = tid & 31;
    const int lane16 = lane & 15;
    const int half   = lane >> 4;            // A/B K-half selector (wave32 layout)
    const int wave   = tid >> 5;

    const int nblocks = Out >> 7;            // N covered: 128 per block
    const int mblk    = blockIdx.x / nblocks;
    const int nblk    = blockIdx.x % nblocks;
    const int m_base  = mblk * 64;
    const int col     = nblk * 128 + wave * 16 + lane16;   // this lane's column

    // producer role: row er (0..63), feature slot efi (0..3) -> exactly 256 thr
    // handles features (i0+efi) -> panel 0 and (i0+4+efi) -> panel 1.
    // A-frag layout: K=fi*8+p ; half=fi&1 ; element block = (fi>>1)*8
    const int er    = tid & 63;
    const int efi   = tid >> 6;
    const int es    = er >> 4;
    const int elane = (efi & 1) * 16 + (er & 15);
    const int eoff  = ((es * 32 + elane) * 2 + (efi >> 1)) * 8;   // f16 elems
    const float* xrow = X + (size_t)(m_base + er) * In;

    const int nStages = In >> 3;             // 8 features (64 K) per stage
    v8f acc[4] = {v8f{}, v8f{}, v8f{}, v8f{}};

    auto produce = [&](int buf, int i0) {
        *(v8h*)&ldsA[buf][0][eoff] = basis8(xrow[i0 + efi]);
        *(v8h*)&ldsA[buf][1][eoff] = basis8(xrow[i0 + 4 + efi]);
    };
    auto consume = [&](int buf, int i0) {
        #pragma unroll
        for (int panel = 0; panel < 2; ++panel) {
            const int ip = i0 + panel * 4;
            // B frag (32x16 f16): lanes 0-15 hold K=0..7,16..23, 16-31 K=8..15,24..31;
            // elems 0..7 / 8..15 are two contiguous 16B runs (coalesced)
            const _Float16* wp0 = W + ((size_t)(ip + half)     * Out + col) * PP;
            const _Float16* wp1 = W + ((size_t)(ip + 2 + half) * Out + col) * PP;
            const v8h b0 = *(const v8h*)wp0;
            const v8h b1 = *(const v8h*)wp1;
            const v16h bfrag = __builtin_shufflevector(b0, b1,
                0, 1, 2, 3, 4, 5, 6, 7, 8, 9, 10, 11, 12, 13, 14, 15);
            v16h afrag[4];
            #pragma unroll
            for (int s = 0; s < 4; ++s)
                afrag[s] = *(const v16h*)&ldsA[buf][panel][(s * 32 + lane) * 16];
            #pragma unroll
            for (int s = 0; s < 4; ++s)
                acc[s] = __builtin_amdgcn_wmma_f32_16x16x32_f16(
                    /*neg_a=*/false, afrag[s], /*neg_b=*/false, bfrag,
                    /*c_mod=*/(short)0, acc[s], /*reuse_a=*/false, /*reuse_b=*/false);
        }
    };

    // software pipeline: prologue produce, steady-state, epilogue consume
    produce(0, 0);
    __syncthreads();
    int i0 = 0, buf = 0;
    for (int st = 0; st < nStages - 1; ++st, i0 += 8, buf ^= 1) {
        produce(buf ^ 1, i0 + 8);
        __builtin_prefetch(W + ((size_t)(i0 + 8 + 2 * half) * Out + col) * PP, 0, 3);
        consume(buf, i0);
        __syncthreads();   // buf consumed & next stage visible
    }
    consume(buf, i0);

    // ---- store: C/D layout: VGPR r -> M = r (lanes 0-15) / 8+r (lanes 16-31)
    #pragma unroll
    for (int s = 0; s < 4; ++s) {
        const int rbase = m_base + s * 16 + half * 8;
        #pragma unroll
        for (int r = 0; r < 8; ++r)
            Y[(size_t)(rbase + r) * Out + col] = acc[s][r];
    }
}

// -------------------------------------------------------------------------
// MinGRU combine (in place): h <- (1-sigmoid(zpre)) * cumsum_t(h) + sigmoid*h
// One thread per (b, d) column; coalesced across d, sequential over t.
// -------------------------------------------------------------------------
__global__ __launch_bounds__(256)
void scan_combine_kernel(float* __restrict__ hbar_ht,      // (B,T,D) in/out
                         const float* __restrict__ zpre)   // (B,T,D)
{
    const int gid = blockIdx.x * blockDim.x + threadIdx.x; // b*D + d
    if (gid >= BB * DD) return;
    const int b = gid / DD;
    const int d = gid % DD;
    float accum = 0.f;
    size_t idx = (size_t)b * TT * DD + d;
    for (int t = 0; t < TT; ++t, idx += DD) {
        const float hb = hbar_ht[idx];
        const float z  = 1.f / (1.f + __expf(-zpre[idx]));
        accum += hb;
        hbar_ht[idx] = (1.f - z) * accum + z * hb;
    }
}

// -------------------------------------------------------------------------
extern "C" void kernel_launch(void* const* d_in, const int* in_sizes, int n_in,
                              void* d_out, int out_size, void* d_ws, size_t ws_size,
                              hipStream_t stream)
{
    const float* x       = (const float*)d_in[0];  // (B,T,IN)
    const float* wz0     = (const float*)d_in[1];  // (IN,D,P)
    const float* wh0     = (const float*)d_in[2];  // (IN,D,P)
    const float* wo0     = (const float*)d_in[3];  // (D,D,P)
    const float* wz_rest = (const float*)d_in[4];  // (L-1,D,D,P)
    const float* wh_rest = (const float*)d_in[5];  // (L-1,D,D,P)
    const float* wo_rest = (const float*)d_in[6];  // (L-1,D,D,P)
    const float* w_out   = (const float*)d_in[7];  // (D,OUT,P)

    float* out_main = (float*)d_out;                       // (B,T,OUT)
    float* hidden   = out_main + (size_t)MM * OUTF;        // (L,B,T,D)

    const size_t actElems = (size_t)MM * DD;               // 32 MB each
    float* bufZ   = (float*)d_ws;
    float* bufCur = bufZ + actElems;

    // f16 weight arena after the two activation buffers
    _Float16* wArena = (_Float16*)(bufCur + actElems);
    const size_t nSmall = (size_t)INF * DD * PP;           // 262,144
    const size_t nBig   = (size_t)DD * DD * PP;            // 2,097,152
    const size_t nHead  = (size_t)DD * OUTF * PP;          // 1,048,576

    _Float16* wh0_16 = wArena;
    _Float16* wz0_16 = wh0_16 + nSmall;
    _Float16* wo0_16 = wz0_16 + nSmall;
    _Float16* whr_16 = wo0_16 + nBig;                      // 3 layers
    _Float16* wzr_16 = whr_16 + 3 * nBig;
    _Float16* wor_16 = wzr_16 + 3 * nBig;
    _Float16* whd_16 = wor_16 + 3 * nBig;                  // head

    auto cvt = [&](const float* src, _Float16* dst, size_t n) {
        const int n8 = (int)(n / 8);
        w_to_f16_kernel<<<(n8 + 255) / 256, 256, 0, stream>>>(src, dst, n8);
    };
    cvt(wh0, wh0_16, nSmall);
    cvt(wz0, wz0_16, nSmall);
    cvt(wo0, wo0_16, nBig);
    cvt(wh_rest, whr_16, 3 * nBig);
    cvt(wz_rest, wzr_16, 3 * nBig);
    cvt(wo_rest, wor_16, 3 * nBig);
    cvt(w_out, whd_16, nHead);

    auto gemm = [&](const float* Xp, const _Float16* Wp, float* Yp, int In, int Out) {
        const int blocks = (MM / 64) * (Out / 128);        // 64x128 tile per block
        apl_gemm_kernel<<<blocks, 256, 0, stream>>>(Xp, Wp, Yp, In, Out);
    };
    auto scan = [&](float* hbar_ht, const float* zpre) {
        scan_combine_kernel<<<(BB * DD + 255) / 256, 256, 0, stream>>>(hbar_ht, zpre);
    };

    // ---- layer 0 (IN=64 -> D) ----
    float* h0 = hidden;                                    // hbar then ht in place
    gemm(x, wh0_16, h0,   INF, DD);
    gemm(x, wz0_16, bufZ, INF, DD);
    scan(h0, bufZ);
    gemm(h0, wo0_16, bufCur, DD, DD);

    // ---- layers 1..L-1 (D -> D) ----
    for (int l = 0; l < LL - 1; ++l) {
        float* hl = hidden + (size_t)(l + 1) * actElems;
        gemm(bufCur, whr_16 + (size_t)l * nBig, hl,   DD, DD);
        gemm(bufCur, wzr_16 + (size_t)l * nBig, bufZ, DD, DD);
        scan(hl, bufZ);
        gemm(hl, wor_16 + (size_t)l * nBig, bufCur, DD, DD);
    }

    // ---- output head (D -> OUT) ----
    gemm(bufCur, whd_16, out_main, DD, OUTF);
}
